// GatingAttentionOpti_29094108463292
// MI455X (gfx1250) — matrix operand order repaired and checked by reading, not compiled, not run on hardware
//
#include <hip/hip_runtime.h>
#include <hip/hip_bf16.h>

typedef _Float16 h16;
typedef __attribute__((ext_vector_type(16))) _Float16 v16h;
typedef __attribute__((ext_vector_type(8)))  _Float16 v8h;
typedef __attribute__((ext_vector_type(8)))  float    v8f;

#define BB   128
#define NQ   384
#define NK   384
#define NH   8
#define AD   256   // model dim (== NUM_HEAD*KEY_DIM)
#define OD   256   // output dim

// ---- operand load helpers (two 16B chunks -> v16h) ----
__device__ __forceinline__ v16h ld16_2(const h16* p0, const h16* p1) {
  v8h a = *(const v8h*)p0;
  v8h b = *(const v8h*)p1;
  v16h r;
#pragma unroll
  for (int i = 0; i < 8; ++i) { r[i] = a[i]; r[i + 8] = b[i]; }
  return r;
}
__device__ __forceinline__ v16h ld16_c(const h16* p) { return ld16_2(p, p + 8); }

__device__ __forceinline__ v8f wmma_f16(v16h A, v16h Bm, v8f C) {
  return __builtin_amdgcn_wmma_f32_16x16x32_f16(false, A, false, Bm, (short)0, C,
                                                false, false);
}

// ============================================================
// Kernel 0: convert weights to f16, transposed so contraction
// dim is contiguous (B-operand layout). Scale baked into qwT.
// ============================================================
__global__ void prep_w(const float* __restrict__ qw, const float* __restrict__ kw,
                       const float* __restrict__ vw, const float* __restrict__ gw,
                       const float* __restrict__ ow,
                       h16* qwT, h16* kwT, h16* vwT, h16* gwT, h16* owT) {
  int idx = blockIdx.x * 256 + threadIdx.x;   // 65536 total
  int n = idx >> 8, a = idx & 255;
  const float scale = 0.17677669529663687f;   // 1/sqrt(32)
  qwT[n * 256 + a] = (h16)(qw[a * 256 + n] * scale);
  kwT[n * 256 + a] = (h16)kw[a * 256 + n];
  vwT[n * 256 + a] = (h16)vw[a * 256 + n];
  gwT[n * 256 + a] = (h16)gw[a * 256 + n];
  owT[n * 256 + a] = (h16)ow[a * 256 + n];    // ow flat [hc][o] -> owT[o][hc]
}

// ============================================================
// Kernel 1: Q projection + sigmoid gate (f16).  16 tokens x 256
// cols per workgroup; A-operands hoisted; K=256 -> 8 WMMAs/tile.
// ============================================================
__global__ void proj_q(const float* __restrict__ x, const h16* __restrict__ qwT,
                       const h16* __restrict__ gwT, const float* __restrict__ gb,
                       h16* __restrict__ qh, h16* __restrict__ gate) {
  __shared__ __align__(16) h16 xs[16 * 256];
  const int tid = threadIdx.x;
  const long tokbase = (long)blockIdx.x * 16;
  for (int i = tid; i < 16 * 256; i += 256) {
    int r = i >> 8, c = i & 255;
    xs[i] = (h16)x[(tokbase + r) * 256 + c];
  }
  __syncthreads();

  const int wave = tid >> 5, lane = tid & 31;
  const int n = lane & 15, hi = lane >> 4;
  const long b = tokbase / NQ;
  const long q0 = tokbase % NQ;

  v16h A[8];
#pragma unroll
  for (int kk = 0; kk < 8; ++kk) {
    const h16* ap = &xs[n * 256 + kk * 32 + hi * 16];
    A[kk] = ld16_c(ap);
  }

#pragma unroll
  for (int i = 0; i < 4; ++i) {
    const int task = wave + 8 * i;
    const int mat = task >> 4;      // 0 = q, 1 = gate
    const int ct  = task & 15;      // column tile
    const h16* wT = mat ? gwT : qwT;
    v8f acc = {};
#pragma unroll
    for (int kk = 0; kk < 8; ++kk) {
      const h16* bp = &wT[(ct * 16 + n) * 256 + kk * 32 + hi * 16];
      acc = wmma_f16(A[kk], ld16_c(bp), acc);
    }
    const int col = ct * 16 + n;           // 0..255
    if (mat == 0) {
      const int h = col >> 5, c32 = col & 31;
#pragma unroll
      for (int r = 0; r < 8; ++r) {
        long q = q0 + r + hi * 8;
        qh[((b * NH + h) * NQ + q) * 32 + c32] = (h16)acc[r];
      }
    } else {
      const float bias = gb[col];
#pragma unroll
      for (int r = 0; r < 8; ++r) {
        long q = q0 + r + hi * 8;
        float v = acc[r] + bias;
        gate[(b * NQ + q) * 256 + col] = (h16)(1.0f / (1.0f + __expf(-v)));
      }
    }
  }
}

// ============================================================
// Kernel 2: K / V projections.  K stored [b][h][k][32];
// V stored transposed [b][h][c][384] for contiguous PV B-loads.
// ============================================================
__global__ void proj_kv(const float* __restrict__ x, const h16* __restrict__ kwT,
                        const h16* __restrict__ vwT, h16* __restrict__ kh,
                        h16* __restrict__ vt) {
  __shared__ __align__(16) h16 xs[16 * 256];
  const int tid = threadIdx.x;
  const long tokbase = (long)blockIdx.x * 16;
  for (int i = tid; i < 16 * 256; i += 256) {
    int r = i >> 8, c = i & 255;
    xs[i] = (h16)x[(tokbase + r) * 256 + c];
  }
  __syncthreads();

  const int wave = tid >> 5, lane = tid & 31;
  const int n = lane & 15, hi = lane >> 4;
  const long b = tokbase / NK;
  const long k0 = tokbase % NK;

  v16h A[8];
#pragma unroll
  for (int kk = 0; kk < 8; ++kk) {
    const h16* ap = &xs[n * 256 + kk * 32 + hi * 16];
    A[kk] = ld16_c(ap);
  }

#pragma unroll
  for (int i = 0; i < 4; ++i) {
    const int task = wave + 8 * i;
    const int mat = task >> 4;      // 0 = k, 1 = v
    const int ct  = task & 15;
    const h16* wT = mat ? vwT : kwT;
    v8f acc = {};
#pragma unroll
    for (int kk = 0; kk < 8; ++kk) {
      const h16* bp = &wT[(ct * 16 + n) * 256 + kk * 32 + hi * 16];
      acc = wmma_f16(A[kk], ld16_c(bp), acc);
    }
    const int col = ct * 16 + n;
    const int h = col >> 5, c32 = col & 31;
    if (mat == 0) {
#pragma unroll
      for (int r = 0; r < 8; ++r) {
        long k = k0 + r + hi * 8;
        kh[((b * NH + h) * NK + k) * 32 + c32] = (h16)acc[r];
      }
    } else {
      v8h pk;
#pragma unroll
      for (int r = 0; r < 8; ++r) pk[r] = (h16)acc[r];
      *(v8h*)&vt[((b * NH + h) * 32 + c32) * (long)NK + k0 + hi * 8] = pk;
    }
  }
}

// ============================================================
// Kernel 3: flash attention per (b,h,128-query block).
// 8 waves x 16 query rows; 12 chunks of 32 keys staged once per
// block into LDS (K by threads 0-127, V by 128-255); online
// softmax; gating fused in epilogue.
// ============================================================
__global__ void attn(const h16* __restrict__ qh, const h16* __restrict__ kh,
                     const h16* __restrict__ vt, const float* __restrict__ bias,
                     const float* __restrict__ nbias, const h16* __restrict__ gate,
                     h16* __restrict__ wah) {
  __shared__ __align__(16) h16 pbuf[8][16 * 32];
  __shared__ __align__(16) h16 ks[32 * 32];   // [key][c]
  __shared__ __align__(16) h16 vs[32 * 32];   // [c][key]
  const int tid = threadIdx.x, wave = tid >> 5, lane = tid & 31;
  const int n = lane & 15, hi = lane >> 4;
  const int bh = blockIdx.x;             // b*8 + h
  const int b = bh >> 3, h = bh & 7;
  const int q0 = blockIdx.y * 128 + wave * 16;

  // Q A-operand (rows = q0..q0+15, K = head dim 32)
  const h16* qp = &qh[((long)bh * NQ + q0 + n) * 32 + hi * 16];
  const v16h Aq = ld16_c(qp);

  const h16* kbase = &kh[(long)bh * NK * 32];
  const h16* vbase = &vt[(long)bh * 32 * NK];
  const float* bptr = &bias[(long)b * NQ * NK];

  const int t2 = tid & 127;
  const int r0 = t2 >> 2, o8 = (t2 & 3) * 8;   // staging coords

  float m[8], l[8];
#pragma unroll
  for (int r = 0; r < 8; ++r) { m[r] = -1.0e30f; l[r] = 0.0f; }
  v8f O0 = {}, O1 = {};
  h16* pb = pbuf[wave];

  for (int kt = 0; kt < NK / 32; ++kt) {
    const int kb = kt * 32;
    // cooperative staging of this 32-key K/V chunk (4 KB total)
    __syncthreads();
    if (tid < 128)
      *(v8h*)&ks[r0 * 32 + o8] = *(const v8h*)&kbase[(long)(kb + r0) * 32 + o8];
    else
      *(v8h*)&vs[r0 * 32 + o8] = *(const v8h*)&vbase[(long)r0 * NK + kb + o8];
    if (kt + 1 < NK / 32) {   // prefetch next chunk (global_prefetch_b8)
      if (tid < 128)
        __builtin_prefetch(&kbase[(long)(kb + 32 + r0) * 32 + o8], 0, 1);
      else
        __builtin_prefetch(&vbase[(long)r0 * NK + kb + 32 + o8], 0, 1);
    }
    __syncthreads();

    // S = Q x K^T  (two 16x16 tiles covering 32 keys)
    v16h B0 = ld16_c(&ks[n * 32 + hi * 16]);
    v16h B1 = ld16_c(&ks[(16 + n) * 32 + hi * 16]);
    v8f zero = {};
    v8f S0 = wmma_f16(Aq, B0, zero);
    v8f S1 = wmma_f16(Aq, B1, zero);
#pragma unroll
    for (int r = 0; r < 8; ++r) {
      long q = q0 + r + hi * 8;
      S0[r] += bptr[q * NK + kb + n]      + nbias[q * NK + kb + n];
      S1[r] += bptr[q * NK + kb + 16 + n] + nbias[q * NK + kb + 16 + n];
    }
    // online softmax update (row stats live per 16-lane group)
#pragma unroll
    for (int r = 0; r < 8; ++r) {
      float v = fmaxf(S0[r], S1[r]);
#pragma unroll
      for (int mask = 1; mask < 16; mask <<= 1)
        v = fmaxf(v, __shfl_xor(v, mask));
      float mn = fmaxf(m[r], v);
      float alpha = __expf(m[r] - mn);
      S0[r] = __expf(S0[r] - mn);
      S1[r] = __expf(S1[r] - mn);
      float s = S0[r] + S1[r];
#pragma unroll
      for (int mask = 1; mask < 16; mask <<= 1)
        s += __shfl_xor(s, mask);
      l[r] = l[r] * alpha + s;
      m[r] = mn;
      O0[r] *= alpha;
      O1[r] *= alpha;
    }
    // stage P (C-layout -> A-layout) through per-wave LDS
#pragma unroll
    for (int r = 0; r < 8; ++r) {
      const int row = r + hi * 8;
      pb[row * 32 + n]      = (h16)S0[r];
      pb[row * 32 + 16 + n] = (h16)S1[r];
    }
    asm volatile("s_wait_dscnt 0" ::: "memory");     // cross-lane exchange visible
    const h16* ap = &pb[n * 32 + hi * 16];
    v16h Ap = ld16_c(ap);
    asm volatile("s_wait_dscnt 0" ::: "memory");     // loads done before overwrite
    // O += P x V (32 keys contracted; V chunk in LDS, contiguous B loads)
    v16h Bv0 = ld16_c(&vs[n * 32 + hi * 16]);
    v16h Bv1 = ld16_c(&vs[(16 + n) * 32 + hi * 16]);
    O0 = wmma_f16(Ap, Bv0, O0);
    O1 = wmma_f16(Ap, Bv1, O1);
  }

  // epilogue: normalize, gate, store f16 [b][q][h*32+c]
  const h16* gptr = &gate[(long)b * NQ * 256];
#pragma unroll
  for (int r = 0; r < 8; ++r) {
    long q = q0 + r + hi * 8;
    float inv = 1.0f / l[r];
    float g0 = (float)gptr[q * 256 + h * 32 + n];
    float g1 = (float)gptr[q * 256 + h * 32 + 16 + n];
    wah[((long)b * NQ + q) * 256 + h * 32 + n]      = (h16)(O0[r] * inv * g0);
    wah[((long)b * NQ + q) * 256 + h * 32 + 16 + n] = (h16)(O1[r] * inv * g1);
  }
}

// ============================================================
// Kernel 4: output projection [49152,256] x [256,256] + bias
// ============================================================
__global__ void outproj(const h16* __restrict__ wah, const h16* __restrict__ owT,
                        const float* __restrict__ ob, float* __restrict__ out) {
  const int tid = threadIdx.x, wave = tid >> 5, lane = tid & 31;
  const int n = lane & 15, hi = lane >> 4;
  const long tokbase = (long)blockIdx.x * 16;

  v16h A[8];
#pragma unroll
  for (int kk = 0; kk < 8; ++kk) {
    const h16* ap = wah + (tokbase + n) * 256 + kk * 32 + hi * 16;
    A[kk] = ld16_c(ap);
  }

#pragma unroll
  for (int i = 0; i < 2; ++i) {
    const int ct = wave * 2 + i;
    v8f acc = {};
#pragma unroll
    for (int kk = 0; kk < 8; ++kk) {
      const h16* bp = owT + (long)(ct * 16 + n) * 256 + kk * 32 + hi * 16;
      acc = wmma_f16(A[kk], ld16_c(bp), acc);
    }
    const int col = ct * 16 + n;
    const float bb = ob[col];
#pragma unroll
    for (int r = 0; r < 8; ++r)
      out[(tokbase + r + hi * 8) * 256 + col] = acc[r] + bb;
  }
}

// ============================================================
extern "C" void kernel_launch(void* const* d_in, const int* in_sizes, int n_in,
                              void* d_out, int out_size, void* d_ws, size_t ws_size,
                              hipStream_t stream) {
  const float* q_data  = (const float*)d_in[0];
  const float* m_data  = (const float*)d_in[1];
  const float* bias    = (const float*)d_in[2];
  const float* nbias   = (const float*)d_in[3];
  const float* query_w = (const float*)d_in[4];
  const float* key_w   = (const float*)d_in[5];
  const float* value_w = (const float*)d_in[6];
  const float* gating_w= (const float*)d_in[7];
  const float* gating_b= (const float*)d_in[8];
  const float* output_w= (const float*)d_in[9];
  const float* output_b= (const float*)d_in[10];
  float* out = (float*)d_out;

  char* ws = (char*)d_ws;
  const size_t WSZ = 256 * 256 * sizeof(h16);          // 128 KB each
  h16* qwT = (h16*)ws;                 ws += WSZ;
  h16* kwT = (h16*)ws;                 ws += WSZ;
  h16* vwT = (h16*)ws;                 ws += WSZ;
  h16* gwT = (h16*)ws;                 ws += WSZ;
  h16* owT = (h16*)ws;                 ws += WSZ;
  const size_t T16 = (size_t)BB * NH * NQ * 32 * sizeof(h16);  // 25.2 MB
  h16* qh = (h16*)ws;                  ws += T16;
  h16* kh = (h16*)ws;                  ws += T16;
  h16* vt = (h16*)ws;                  ws += T16;
  h16* gate = (h16*)ws;                ws += (size_t)BB * NQ * 256 * sizeof(h16);
  h16* wah = (h16*)ws;                 ws += (size_t)BB * NQ * 256 * sizeof(h16);

  prep_w<<<256, 256, 0, stream>>>(query_w, key_w, value_w, gating_w, output_w,
                                  qwT, kwT, vwT, gwT, owT);
  proj_q<<<BB * NQ / 16, 256, 0, stream>>>(q_data, qwT, gwT, gating_b, qh, gate);
  proj_kv<<<BB * NK / 16, 256, 0, stream>>>(m_data, kwT, vwT, kh, vt);
  attn<<<dim3(BB * NH, NQ / 128), 256, 0, stream>>>(qh, kh, vt, bias, nbias, gate, wah);
  outproj<<<BB * NQ / 16, 256, 0, stream>>>(wah, owT, output_b, out);
}
